// IntSequenceRNNDecoder_36584531427445
// MI455X (gfx1250) — matrix operand order repaired and checked by reading, not compiled, not run on hardware
//
#include <hip/hip_runtime.h>
#include <stdint.h>
#include <stddef.h>

// Problem constants (match reference)
#define B_    1024
#define IN_   512
#define LEN_  64
#define V_    1024
#define E_    256
#define H_    1024
#define G_    3072   // 3*H

typedef __attribute__((ext_vector_type(8)))  float   v8f;
typedef __attribute__((ext_vector_type(8)))  __bf16  v8bf;
typedef __attribute__((ext_vector_type(16))) __bf16  v16bf;
typedef __attribute__((ext_vector_type(4)))  int     v4i;

// gfx1250 async global->LDS path (probe via __has_builtin; fall back to sync staging)
#if __has_builtin(__builtin_amdgcn_global_load_async_to_lds_b128)
#define HAVE_ASYNC_LDS 1
#else
#define HAVE_ASYNC_LDS 0
#endif

// Builtin signature (from clang diagnostic): (v4i AS1*, v4i AS3*, Imm offset, Imm cpol)
#define ASG(p) ((__attribute__((address_space(1))) v4i*)((v4i*)(void*)(p)))
#define ASL(p) ((__attribute__((address_space(3))) v4i*)((v4i*)(void*)(p)))

// ---------------------------------------------------------------------------
// fp32 -> bf16 conversion
// ---------------------------------------------------------------------------
__global__ __launch_bounds__(256) void cvt_f32_bf16_kernel(
    const float* __restrict__ in, __bf16* __restrict__ out, int n) {
  int i = blockIdx.x * 256 + threadIdx.x;
  if (i < n) out[i] = (__bf16)in[i];
}

// ---------------------------------------------------------------------------
// C[M,N] = X[M,K](bf16) * W[N,K]^T(bf16) + bias[N]   (fp32 out, ldc row stride)
// 128x128 block tile, 8 waves (2x4), wave tile 64x32 -> 4x2 WMMA accumulators.
// K-loop step 32 with double-buffered LDS staging (async-to-LDS if available).
// ---------------------------------------------------------------------------
__global__ __launch_bounds__(256) void gemm_bf16_wmma_kernel(
    const __bf16* __restrict__ X, const __bf16* __restrict__ W,
    const float* __restrict__ bias, float* __restrict__ C,
    int M, int N, int K, int ldc) {
  constexpr int LSTR = 56;              // halfs per LDS row: 112B, 16B aligned, conflict-free
  __shared__ __bf16 Xs[2][128 * LSTR];
  __shared__ __bf16 Ws[2][128 * LSTR];

  const int tid  = threadIdx.x;
  const int lane = tid & 31;
  const int wave = tid >> 5;
  const int wm = wave >> 2;             // 0..1  : wave M tile (64 rows)
  const int wn = wave & 3;              // 0..3  : wave N tile (32 cols)
  const int fm = lane & 15;             // row/col within a 16-wide fragment
  const int fh = lane >> 4;             // K-half selector

  const int m0 = blockIdx.y * 128;
  const int n0 = blockIdx.x * 128;

  // staging: 256 threads cover 128 rows x 32 halfs (2 threads/row x 16 halfs)
  const int srow = tid >> 1;
  const int sseg = (tid & 1) * 16;

  const __bf16* Xg = X + (size_t)(m0 + srow) * K + sseg;
  const __bf16* Wg = W + (size_t)(n0 + srow) * K + sseg;
  const int lstage = srow * LSTR + sseg;

  auto stage = [&](int kt, int buf) {
    const __bf16* gx = Xg + kt * 32;
    const __bf16* gw = Wg + kt * 32;
    __bf16* lx = &Xs[buf][lstage];
    __bf16* lw = &Ws[buf][lstage];
#if HAVE_ASYNC_LDS
    __builtin_amdgcn_global_load_async_to_lds_b128(ASG(gx),     ASL(lx),     0, 0);
    __builtin_amdgcn_global_load_async_to_lds_b128(ASG(gx + 8), ASL(lx + 8), 0, 0);
    __builtin_amdgcn_global_load_async_to_lds_b128(ASG(gw),     ASL(lw),     0, 0);
    __builtin_amdgcn_global_load_async_to_lds_b128(ASG(gw + 8), ASL(lw + 8), 0, 0);
#else
    v8bf x0 = *(const v8bf*)(gx);
    v8bf x1 = *(const v8bf*)(gx + 8);
    v8bf w0 = *(const v8bf*)(gw);
    v8bf w1 = *(const v8bf*)(gw + 8);
    *(v8bf*)(lx)     = x0;  *(v8bf*)(lx + 8) = x1;
    *(v8bf*)(lw)     = w0;  *(v8bf*)(lw + 8) = w1;
#endif
  };

  auto wait_stage = [&]() {
#if HAVE_ASYNC_LDS
    asm volatile("s_wait_asynccnt 0" ::: "memory");
#else
    asm volatile("s_wait_dscnt 0" ::: "memory");
#endif
  };

  const v8f vzero = {0.f, 0.f, 0.f, 0.f, 0.f, 0.f, 0.f, 0.f};
  v8f acc[4][2];
#pragma unroll
  for (int mf = 0; mf < 4; ++mf)
#pragma unroll
    for (int nf = 0; nf < 2; ++nf) acc[mf][nf] = vzero;

  const int nk = K >> 5;
  stage(0, 0);
  wait_stage();
  __syncthreads();

  for (int kt = 0; kt < nk; ++kt) {
    const int cur = kt & 1;

    // A fragments: lane fm holds row M=fm; fh selects K chunks {fh*8.. , 16+fh*8..}
    v16bf a[4];
#pragma unroll
    for (int mf = 0; mf < 4; ++mf) {
      const __bf16* p = &Xs[cur][(wm * 64 + mf * 16 + fm) * LSTR];
      v8bf lo = *(const v8bf*)(p + fh * 8);
      v8bf hi = *(const v8bf*)(p + 16 + fh * 8);
      a[mf] = __builtin_shufflevector(lo, hi, 0, 1, 2, 3, 4, 5, 6, 7,
                                      8, 9, 10, 11, 12, 13, 14, 15);
    }
    // B fragments: lane fm holds column N=fm; fh selects K chunks {fh*8.. , 16+fh*8..}
    v16bf b[2];
#pragma unroll
    for (int nf = 0; nf < 2; ++nf) {
      const __bf16* p = &Ws[cur][(wn * 32 + nf * 16 + fm) * LSTR];
      v8bf lo = *(const v8bf*)(p + fh * 8);
      v8bf hi = *(const v8bf*)(p + 16 + fh * 8);
      b[nf] = __builtin_shufflevector(lo, hi, 0, 1, 2, 3, 4, 5, 6, 7,
                                      8, 9, 10, 11, 12, 13, 14, 15);
    }

    // fragments are in VGPRs -> release this buffer for the next stage
    asm volatile("s_wait_dscnt 0" ::: "memory");
    __syncthreads();
    if (kt + 1 < nk) stage(kt + 1, cur ^ 1);

#pragma unroll
    for (int mf = 0; mf < 4; ++mf)
#pragma unroll
      for (int nf = 0; nf < 2; ++nf)
        acc[mf][nf] = __builtin_amdgcn_wmma_f32_16x16x32_bf16(
            false, a[mf], false, b[nf], (short)0, acc[mf][nf], false, false);

    if (kt + 1 < nk) { wait_stage(); __syncthreads(); }
  }

  // Epilogue. C/D layout: N = lane&15, M = vgpr + (lane>=16 ? 8 : 0)
#pragma unroll
  for (int nf = 0; nf < 2; ++nf) {
    const int col = n0 + wn * 32 + nf * 16 + fm;
    const float bv = bias ? bias[col] : 0.0f;
#pragma unroll
    for (int mf = 0; mf < 4; ++mf) {
      const int rbase = m0 + wm * 64 + mf * 16 + fh * 8;
#pragma unroll
      for (int r = 0; r < 8; ++r)
        C[(size_t)(rbase + r) * ldc + col] = acc[mf][nf][r] + bv;
    }
  }
}

// ---------------------------------------------------------------------------
// GRU gate fusion: h = (1-z)*n + z*h  (PyTorch gate order r,z,n)
// gi/gh are [B, 3H]; h fp32 in/out + bf16 copy for the next GEMM.
// ---------------------------------------------------------------------------
__global__ __launch_bounds__(256) void gru_gate_kernel(
    const float* __restrict__ gi, const float* __restrict__ gh,
    float* __restrict__ h, __bf16* __restrict__ hb) {
  int i = blockIdx.x * 256 + threadIdx.x;     // over B_*H_
  int bb = i >> 10;                           // H_ == 1024
  int j  = i & (H_ - 1);
  size_t g = (size_t)bb * G_ + j;
  float r = 1.0f / (1.0f + __expf(-(gi[g] + gh[g])));
  float z = 1.0f / (1.0f + __expf(-(gi[g + H_] + gh[g + H_])));
  float n = tanhf(gi[g + 2 * H_] + r * gh[g + 2 * H_]);
  float hv = (1.0f - z) * n + z * h[i];
  h[i]  = hv;
  hb[i] = (__bf16)hv;
}

// ---------------------------------------------------------------------------
// init: h0 = h1 = pre-linear output (fp32 + bf16), x0 = broadcast(sos)
// ---------------------------------------------------------------------------
__global__ __launch_bounds__(256) void init_state_kernel(
    const float* __restrict__ hsrc, float* __restrict__ h0, float* __restrict__ h1,
    __bf16* __restrict__ h0b, __bf16* __restrict__ h1b,
    const __bf16* __restrict__ sosb, __bf16* __restrict__ xb) {
  int i = blockIdx.x * 256 + threadIdx.x;     // over B_*H_
  float v = hsrc[i];
  h0[i] = v; h1[i] = v;
  __bf16 vb = (__bf16)v;
  h0b[i] = vb; h1b[i] = vb;
  if (i < B_ * E_) xb[i] = sosb[i & (E_ - 1)];
}

// ---------------------------------------------------------------------------
// Per-row argmax over V logits (first-max tie-break) + embedding gather (bf16)
// ---------------------------------------------------------------------------
__global__ __launch_bounds__(256) void argmax_embed_kernel(
    const float* __restrict__ logits, const __bf16* __restrict__ embb,
    int* __restrict__ out_idx, __bf16* __restrict__ xb, int t) {
  __shared__ float sv[256];
  __shared__ int   si[256];
  const int b = blockIdx.x;
  const float* row = logits + (size_t)b * (LEN_ * V_) + (size_t)t * V_;
  float best = -3.4e38f; int bi = 0;
  for (int v = threadIdx.x; v < V_; v += 256) {
    float x = row[v];
    if (x > best) { best = x; bi = v; }
  }
  sv[threadIdx.x] = best; si[threadIdx.x] = bi;
  __syncthreads();
  for (int s = 128; s > 0; s >>= 1) {
    if ((int)threadIdx.x < s) {
      float ov = sv[threadIdx.x + s]; int oi = si[threadIdx.x + s];
      if (ov > sv[threadIdx.x] ||
          (ov == sv[threadIdx.x] && oi < si[threadIdx.x])) {
        sv[threadIdx.x] = ov; si[threadIdx.x] = oi;
      }
    }
    __syncthreads();
  }
  const int idx = si[0];
  if (threadIdx.x == 0) out_idx[(size_t)b * LEN_ + t] = idx;
  xb[(size_t)b * E_ + threadIdx.x] = embb[(size_t)idx * E_ + threadIdx.x];  // E_ == 256
}

// ---------------------------------------------------------------------------
// Host driver: stream-ordered launch chain (graph-capture safe)
// ---------------------------------------------------------------------------
extern "C" void kernel_launch(void* const* d_in, const int* in_sizes, int n_in,
                              void* d_out, int out_size, void* d_ws, size_t ws_size,
                              hipStream_t stream) {
  (void)in_sizes; (void)n_in; (void)out_size; (void)ws_size;

  const float* latent  = (const float*)d_in[0];
  const float* embed_W = (const float*)d_in[1];
  const float* pre_W   = (const float*)d_in[2];
  const float* pre_b   = (const float*)d_in[3];
  const float* pro_W   = (const float*)d_in[4];
  const float* pro_b   = (const float*)d_in[5];
  const float* w_ih0   = (const float*)d_in[6];
  const float* w_hh0   = (const float*)d_in[7];
  const float* b_ih0   = (const float*)d_in[8];
  const float* b_hh0   = (const float*)d_in[9];
  const float* w_ih1   = (const float*)d_in[10];
  const float* w_hh1   = (const float*)d_in[11];
  const float* b_ih1   = (const float*)d_in[12];
  const float* b_hh1   = (const float*)d_in[13];
  const float* sos     = (const float*)d_in[14];

  // Output: [int32 argmax B*LEN] ++ [fp32 logits B*LEN*V]
  int*   out_idx    = (int*)d_out;
  float* out_logits = (float*)((char*)d_out + (size_t)B_ * LEN_ * sizeof(int));

  char* ws = (char*)d_ws;
  auto alloc = [&](size_t nbytes) {
    char* p = ws; ws += (nbytes + 255) & ~(size_t)255; return p;
  };
  __bf16* latB  = (__bf16*)alloc((size_t)B_ * IN_ * 2);
  __bf16* preWB = (__bf16*)alloc((size_t)H_ * IN_ * 2);
  __bf16* proWB = (__bf16*)alloc((size_t)V_ * H_ * 2);
  __bf16* wih0B = (__bf16*)alloc((size_t)G_ * E_ * 2);
  __bf16* whh0B = (__bf16*)alloc((size_t)G_ * H_ * 2);
  __bf16* wih1B = (__bf16*)alloc((size_t)G_ * H_ * 2);
  __bf16* whh1B = (__bf16*)alloc((size_t)G_ * H_ * 2);
  __bf16* embB  = (__bf16*)alloc((size_t)V_ * E_ * 2);
  __bf16* sosB  = (__bf16*)alloc((size_t)E_ * 2);
  __bf16* xB    = (__bf16*)alloc((size_t)B_ * E_ * 2);
  __bf16* h0B   = (__bf16*)alloc((size_t)B_ * H_ * 2);
  __bf16* h1B   = (__bf16*)alloc((size_t)B_ * H_ * 2);
  float*  h0f   = (float*)alloc((size_t)B_ * H_ * 4);
  float*  h1f   = (float*)alloc((size_t)B_ * H_ * 4);
  float*  giW   = (float*)alloc((size_t)B_ * G_ * 4);
  float*  ghW   = (float*)alloc((size_t)B_ * G_ * 4);

  auto cvt = [&](const float* in, __bf16* o, int n) {
    cvt_f32_bf16_kernel<<<(n + 255) / 256, 256, 0, stream>>>(in, o, n);
  };
  cvt(latent,  latB,  B_ * IN_);
  cvt(pre_W,   preWB, H_ * IN_);
  cvt(pro_W,   proWB, V_ * H_);
  cvt(w_ih0,   wih0B, G_ * E_);
  cvt(w_hh0,   whh0B, G_ * H_);
  cvt(w_ih1,   wih1B, G_ * H_);
  cvt(w_hh1,   whh1B, G_ * H_);
  cvt(embed_W, embB,  V_ * E_);
  cvt(sos,     sosB,  E_);

  auto gemm = [&](const __bf16* X, const __bf16* W, const float* bias, float* C,
                  int M, int N, int K, int ldc) {
    gemm_bf16_wmma_kernel<<<dim3(N / 128, M / 128), 256, 0, stream>>>(
        X, W, bias, C, M, N, K, ldc);
  };

  // pre-linear: h = latent @ pre_W^T + pre_b  (reuse giW as temp)
  gemm(latB, preWB, pre_b, giW, B_, H_, IN_, H_);
  init_state_kernel<<<(B_ * H_) / 256, 256, 0, stream>>>(
      giW, h0f, h1f, h0B, h1B, sosB, xB);

  for (int t = 0; t < LEN_; ++t) {
    // layer 0
    gemm(xB,  wih0B, b_ih0, giW, B_, G_, E_, G_);
    gemm(h0B, whh0B, b_hh0, ghW, B_, G_, H_, G_);
    gru_gate_kernel<<<(B_ * H_) / 256, 256, 0, stream>>>(giW, ghW, h0f, h0B);
    // layer 1
    gemm(h0B, wih1B, b_ih1, giW, B_, G_, H_, G_);
    gemm(h1B, whh1B, b_hh1, ghW, B_, G_, H_, G_);
    gru_gate_kernel<<<(B_ * H_) / 256, 256, 0, stream>>>(giW, ghW, h1f, h1B);
    // projection -> logits written straight into d_out at [b, t, :]
    gemm(h1B, proWB, pro_b, out_logits + (size_t)t * V_, B_, V_, H_, LEN_ * V_);
    // greedy decode + next input embedding
    argmax_embed_kernel<<<B_, 256, 0, stream>>>(out_logits, embB, out_idx, xB, t);
  }
}
